// EncoderLayer_69063074120398
// MI455X (gfx1250) — matrix operand order repaired and checked by reading, not compiled, not run on hardware
//
#include <hip/hip_runtime.h>

// ---------------------------------------------------------------------------
// Types for CDNA5 WMMA (wave32)
// ---------------------------------------------------------------------------
typedef __attribute__((ext_vector_type(16))) __bf16 bf16x16;
typedef __attribute__((ext_vector_type(8)))  __bf16 bf16x8;
typedef __attribute__((ext_vector_type(8)))  float  f32x8;

static __device__ __forceinline__ bf16x16 join8(bf16x8 lo, bf16x8 hi) {
    return __builtin_shufflevector(lo, hi, 0,1,2,3,4,5,6,7,8,9,10,11,12,13,14,15);
}

#define B_   8
#define S_   2048
#define D_   512
#define OUT_ 10

// Async global->LDS copy of 16B per lane (GV mode), tracked by ASYNCcnt.
static __device__ __forceinline__ void async_cp_b128(unsigned ldsOff,
                                                     unsigned long long gaddr) {
    asm volatile("global_load_async_to_lds_b128 %0, %1, off"
                 :
                 : "v"(ldsOff), "v"(gaddr)
                 : "memory");
}

// s_wait_asynccnt needs an immediate operand -> compile-time constant N.
template <int N>
static __device__ __forceinline__ void wait_async_le() {
#if defined(__has_builtin) && __has_builtin(__builtin_amdgcn_s_wait_asynccnt)
    __builtin_amdgcn_s_wait_asynccnt((unsigned short)N);
#else
    asm volatile("s_wait_asynccnt %0" :: "i"(N) : "memory");
#endif
}

// ---------------------------------------------------------------------------
// fp32 -> bf16 convert
// ---------------------------------------------------------------------------
__global__ void k_cvt_bf16(const float* __restrict__ in, __bf16* __restrict__ out, int n) {
    for (int i = blockIdx.x * blockDim.x + threadIdx.x; i < n; i += gridDim.x * blockDim.x)
        out[i] = (__bf16)in[i];
}

// ---------------------------------------------------------------------------
// Transpose + convert: out[c][r] = (bf16) in[r][c], in is [R][C] fp32
// block (32,8), grid (C/32, R/32)
// ---------------------------------------------------------------------------
__global__ void k_tr_cvt(const float* __restrict__ in, __bf16* __restrict__ out, int R, int C) {
    __shared__ float t[32][33];
    const int tx = threadIdx.x, ty = threadIdx.y;
    const int c0 = blockIdx.x * 32, r0 = blockIdx.y * 32;
#pragma unroll
    for (int j = 0; j < 4; ++j)
        t[ty + j * 8][tx] = in[(size_t)(r0 + ty + j * 8) * C + (c0 + tx)];
    __syncthreads();
#pragma unroll
    for (int j = 0; j < 4; ++j)
        out[(size_t)(c0 + ty + j * 8) * R + (r0 + tx)] = (__bf16)t[tx][ty + j * 8];
}

// ---------------------------------------------------------------------------
// bf16 tile transpose with leading dims + batch strides (for V -> V^T)
// ---------------------------------------------------------------------------
__global__ void k_tr_bf16(const __bf16* __restrict__ in, __bf16* __restrict__ out,
                          int ldin, int ldout, long long sIn, long long sOut) {
    __shared__ __bf16 t[32][34];
    in  += (size_t)blockIdx.z * sIn;
    out += (size_t)blockIdx.z * sOut;
    const int tx = threadIdx.x, ty = threadIdx.y;
    const int c0 = blockIdx.x * 32, r0 = blockIdx.y * 32;
#pragma unroll
    for (int j = 0; j < 4; ++j)
        t[ty + j * 8][tx] = in[(size_t)(r0 + ty + j * 8) * ldin + (c0 + tx)];
    __syncthreads();
#pragma unroll
    for (int j = 0; j < 4; ++j)
        out[(size_t)(c0 + ty + j * 8) * ldout + (r0 + tx)] = t[tx][ty + j * 8];
}

// ---------------------------------------------------------------------------
// NT GEMM via V_WMMA_F32_16X16X32_BF16 with async double-buffered LDS staging:
//   C[M][N] (+epilogue) = A[M][K] * Bt[N][K]^T
// block = 256 threads (8 waves), tile 128x128, K-step 32.
// Waves 2(row)x4(col); each wave owns a 64x32 sub-tile = 8 WMMA / K-step.
// Tile fill uses GLOBAL_LOAD_ASYNC_TO_LDS_B128 (ASYNCcnt), double buffered so
// tile t+1 DMA overlaps tile t WMMAs. All M,N %128==0, K %32==0.
// EPI: 0 = f32 store, 1 = bf16 store, 2 = f32 (+bias +residual),
//      3 = bf16 relu(+bias)
// ---------------------------------------------------------------------------
template <int EPI>
__global__ void k_gemm(const __bf16* __restrict__ A, const __bf16* __restrict__ Bt,
                       void* __restrict__ Cv,
                       const float* __restrict__ bias, const float* __restrict__ res,
                       int M, int N, int K, int lda, int ldb, int ldc,
                       long long sA_, long long sB_, long long sC_) {
    constexpr int TKP = 40;                      // 32 + 8 pad (bf16 elems)
    constexpr unsigned BUFB = 128u * TKP * 2u;   // bytes per LDS buffer
    __shared__ __bf16 sA[2][128][TKP];
    __shared__ __bf16 sB[2][128][TKP];

    A  += (size_t)blockIdx.z * sA_;
    Bt += (size_t)blockIdx.z * sB_;
    char* Cb = (char*)Cv;
    const size_t cElem = (EPI == 0 || EPI == 2) ? 4 : 2;
    Cb += (size_t)blockIdx.z * sC_ * cElem;

    const int tid  = threadIdx.x;
    const int lane = tid & 31;
    const int wave = tid >> 5;
    const int row0 = (wave >> 2) * 64;        // wave sub-tile origin in M
    const int col0 = (wave & 3) * 32;         // wave sub-tile origin in N
    const int lm   = lane & 15;
    const int lk8  = (lane >> 4) * 8;         // unused (A and B both use lk16)
    const int lk16 = (lane >> 4) * 16;        // B-fragment K offset
    const int m0   = blockIdx.y * 128;
    const int n0   = blockIdx.x * 128;

    // Per-thread tile-fill assignment: rows r0 and r0+64, fixed 8-elem chunk c0.
    const int r0 = tid >> 2;                  // 0..63
    const int c0 = (tid & 3) * 8;             // 0,8,16,24 (bf16 elems)
    const unsigned long long gA0 =
        (unsigned long long)(uintptr_t)(A + (size_t)(m0 + r0) * lda + c0);
    const unsigned long long gA1 = gA0 + (unsigned long long)64 * lda * 2;
    const unsigned long long gB0 =
        (unsigned long long)(uintptr_t)(Bt + (size_t)(n0 + r0) * ldb + c0);
    const unsigned long long gB1 = gB0 + (unsigned long long)64 * ldb * 2;
    const unsigned lA0 = (unsigned)(uintptr_t)&sA[0][r0][c0];
    const unsigned lA1 = (unsigned)(uintptr_t)&sA[0][r0 + 64][c0];
    const unsigned lB0 = (unsigned)(uintptr_t)&sB[0][r0][c0];
    const unsigned lB1 = (unsigned)(uintptr_t)&sB[0][r0 + 64][c0];

    f32x8 acc[4][2];
    const f32x8 fzero = {0.f,0.f,0.f,0.f,0.f,0.f,0.f,0.f};
#pragma unroll
    for (int i = 0; i < 4; ++i)
#pragma unroll
        for (int j = 0; j < 2; ++j) acc[i][j] = fzero;

    // Prefetch tile 0 into buffer 0 (4 async instructions per wave).
    async_cp_b128(lA0, gA0);
    async_cp_b128(lA1, gA1);
    async_cp_b128(lB0, gB0);
    async_cp_b128(lB1, gB1);

    int buf = 0;
    for (int k0 = 0; k0 < K; k0 += 32) {
        const bool hasNext = (k0 + 32) < K;
        if (hasNext) {
            const unsigned long long go = (unsigned long long)(k0 + 32) * 2;
            const unsigned       lo = (unsigned)(buf ^ 1) * BUFB;
            async_cp_b128(lA0 + lo, gA0 + go);
            async_cp_b128(lA1 + lo, gA1 + go);
            async_cp_b128(lB0 + lo, gB0 + go);
            async_cp_b128(lB1 + lo, gB1 + go);
            wait_async_le<4>();   // previous tile's 4 loads complete (in-order)
        } else {
            wait_async_le<0>();
        }
        __syncthreads();

        bf16x16 af[4];
#pragma unroll
        for (int i = 0; i < 4; ++i) {
            const int r = row0 + i * 16 + lm;
            af[i] = join8(*(const bf16x8*)&sA[buf][r][lk16],
                          *(const bf16x8*)&sA[buf][r][lk16 + 8]);
        }
        bf16x16 bfr[2];
#pragma unroll
        for (int j = 0; j < 2; ++j) {
            const int r = col0 + j * 16 + lm;
            bfr[j] = join8(*(const bf16x8*)&sB[buf][r][lk16],
                           *(const bf16x8*)&sB[buf][r][lk16 + 8]);
        }
#pragma unroll
        for (int i = 0; i < 4; ++i)
#pragma unroll
            for (int j = 0; j < 2; ++j)
                acc[i][j] = __builtin_amdgcn_wmma_f32_16x16x32_bf16(
                    false, af[i], false, bfr[j], (short)0, acc[i][j], false, false);
        __syncthreads();   // all waves done reading buf before it is refilled
        buf ^= 1;
    }

    // Epilogue: C tile 16x16; VGPR e: lanes 0-15 -> M=e, lanes 16-31 -> M=8+e
    const int lr0 = (lane >> 4) * 8;
#pragma unroll
    for (int i = 0; i < 4; ++i) {
#pragma unroll
        for (int j = 0; j < 2; ++j) {
            const int gn = n0 + col0 + j * 16 + lm;
#pragma unroll
            for (int e = 0; e < 8; ++e) {
                const int gm = m0 + row0 + i * 16 + lr0 + e;
                float v = acc[i][j][e];
                const size_t idx = (size_t)gm * ldc + gn;
                if constexpr (EPI == 0) {
                    ((float*)Cb)[idx] = v;
                } else if constexpr (EPI == 1) {
                    ((__bf16*)Cb)[idx] = (__bf16)v;
                } else if constexpr (EPI == 2) {
                    ((float*)Cb)[idx] = v + bias[gn] + res[idx];
                } else {
                    v += bias[gn];
                    ((__bf16*)Cb)[idx] = (__bf16)fmaxf(v, 0.0f);
                }
            }
        }
    }
}

// ---------------------------------------------------------------------------
// Row softmax over L=2048 with scale; f32 in -> bf16 out. 256 thr/row.
// ---------------------------------------------------------------------------
__global__ void k_softmax(const float* __restrict__ in, __bf16* __restrict__ out,
                          int L, float scale) {
    __shared__ float red[256];
    const int tid = threadIdx.x;
    const float* r = in + (size_t)blockIdx.x * L;
    __bf16* o = out + (size_t)blockIdx.x * L;

    float v[8];
    float mx = -3.4e38f;
#pragma unroll
    for (int i = 0; i < 8; ++i) { v[i] = r[tid + i * 256] * scale; mx = fmaxf(mx, v[i]); }
    red[tid] = mx; __syncthreads();
    for (int s = 128; s > 0; s >>= 1) {
        if (tid < s) red[tid] = fmaxf(red[tid], red[tid + s]);
        __syncthreads();
    }
    mx = red[0]; __syncthreads();

    float sum = 0.0f;
#pragma unroll
    for (int i = 0; i < 8; ++i) { v[i] = __expf(v[i] - mx); sum += v[i]; }
    red[tid] = sum; __syncthreads();
    for (int s = 128; s > 0; s >>= 1) {
        if (tid < s) red[tid] += red[tid + s];
        __syncthreads();
    }
    const float inv = 1.0f / red[0];
#pragma unroll
    for (int i = 0; i < 8; ++i) o[tid + i * 256] = (__bf16)(v[i] * inv);
}

// ---------------------------------------------------------------------------
// LayerNorm over rows of 512; one wave32 per row, 8 rows per block.
// ---------------------------------------------------------------------------
__global__ void k_layernorm(const float* __restrict__ in, float* __restrict__ outf,
                            __bf16* __restrict__ outb,
                            const float* __restrict__ g, const float* __restrict__ bb) {
    const int lane = threadIdx.x & 31;
    const int wave = threadIdx.x >> 5;
    const size_t row = (size_t)blockIdx.x * 8 + wave;
    const float* x = in + row * 512;

    float v[16];
    float s = 0.0f;
#pragma unroll
    for (int i = 0; i < 16; ++i) { v[i] = x[lane + i * 32]; s += v[i]; }
    for (int off = 16; off > 0; off >>= 1) s += __shfl_xor(s, off, 32);
    const float mu = s * (1.0f / 512.0f);

    float q = 0.0f;
#pragma unroll
    for (int i = 0; i < 16; ++i) { const float d = v[i] - mu; q += d * d; }
    for (int off = 16; off > 0; off >>= 1) q += __shfl_xor(q, off, 32);
    const float rs = rsqrtf(q * (1.0f / 512.0f) + 1e-5f);

#pragma unroll
    for (int i = 0; i < 16; ++i) {
        const int c = lane + i * 32;
        const float o = (v[i] - mu) * rs * g[c] + bb[c];
        outf[row * 512 + c] = o;
        if (outb) outb[row * 512 + c] = (__bf16)o;
    }
}

// ---------------------------------------------------------------------------
// Mean-pool over S then 10-way classifier. One block (512 thr) per batch.
// ---------------------------------------------------------------------------
__global__ void k_pool_cls(const float* __restrict__ src, const float* __restrict__ wc,
                           const float* __restrict__ bc, float* __restrict__ logits) {
    __shared__ float pooled[512];
    const int b = blockIdx.x, d = threadIdx.x;
    const float* p = src + (size_t)b * S_ * D_ + d;
    float s = 0.0f;
    for (int t = 0; t < S_; ++t) s += p[(size_t)t * D_];
    pooled[d] = s * (1.0f / (float)S_);
    __syncthreads();
    if (d < OUT_) {
        float a = bc[d];
        for (int k = 0; k < D_; ++k) a += pooled[k] * wc[k * OUT_ + d];
        logits[b * OUT_ + d] = a;
    }
}

// ---------------------------------------------------------------------------
// Host-side orchestration
// ---------------------------------------------------------------------------
extern "C" void kernel_launch(void* const* d_in, const int* in_sizes, int n_in,
                              void* d_out, int out_size, void* d_ws, size_t ws_size,
                              hipStream_t stream) {
    const float* x    = (const float*)d_in[0];
    const float* wq   = (const float*)d_in[1];
    const float* wk   = (const float*)d_in[2];
    const float* wv   = (const float*)d_in[3];
    const float* wo   = (const float*)d_in[4];
    const float* bo   = (const float*)d_in[5];
    const float* w1   = (const float*)d_in[6];
    const float* b1   = (const float*)d_in[7];
    const float* w2   = (const float*)d_in[8];
    const float* b2   = (const float*)d_in[9];
    const float* ln1g = (const float*)d_in[10];
    const float* ln1b = (const float*)d_in[11];
    const float* ln2g = (const float*)d_in[12];
    const float* ln2b = (const float*)d_in[13];
    const float* wc   = (const float*)d_in[14];
    const float* bc   = (const float*)d_in[15];

    float* out_logits = (float*)d_out;
    float* out_src    = (float*)d_out + (size_t)B_ * OUT_;   // [B,S,D] fp32

    const int BS = B_ * S_;                 // 16384 rows
    size_t off = 0;
    auto bump = [&](size_t bytes) -> void* {
        void* p = (char*)d_ws + off;
        off += (bytes + 255) & ~(size_t)255;
        return p;
    };

    __bf16* xb    = (__bf16*)bump((size_t)BS * D_ * 2);            // x bf16
    __bf16* Wt    = (__bf16*)bump((size_t)3 * D_ * D_ * 2);        // [wq^T; wk^T; wv^T]
    __bf16* wot   = (__bf16*)bump((size_t)D_ * D_ * 2);
    __bf16* w1t   = (__bf16*)bump((size_t)D_ * D_ * 2);
    __bf16* w2t   = (__bf16*)bump((size_t)D_ * D_ * 2);
    __bf16* QKV   = (__bf16*)bump((size_t)BS * 3 * D_ * 2);        // [16384][1536]
    __bf16* Vt    = (__bf16*)bump((size_t)B_ * D_ * S_ * 2);       // [B][512][2048]
    float*  scores= (float*) bump((size_t)B_ * S_ * S_ * 4);       // [B][2048][2048]
    __bf16* attn  = (__bf16*)bump((size_t)B_ * S_ * S_ * 2);
    __bf16* ctx   = (__bf16*)bump((size_t)BS * D_ * 2);
    float*  t1    = (float*) bump((size_t)BS * D_ * 4);            // x + attn_out
    float*  src1f = (float*) bump((size_t)BS * D_ * 4);
    __bf16* src1b = (__bf16*)bump((size_t)BS * D_ * 2);
    __bf16* h     = (__bf16*)bump((size_t)BS * D_ * 2);
    float*  t2    = (float*) bump((size_t)BS * D_ * 4);            // src1 + ffn_out

    // 1) convert x, transpose-convert weights to bf16 [out][in]
    k_cvt_bf16<<<4096, 256, 0, stream>>>(x, xb, BS * D_);
    dim3 trb(32, 8), trg(16, 16);
    k_tr_cvt<<<trg, trb, 0, stream>>>(wq, Wt,                 D_, D_);
    k_tr_cvt<<<trg, trb, 0, stream>>>(wk, Wt + (size_t)D_*D_, D_, D_);
    k_tr_cvt<<<trg, trb, 0, stream>>>(wv, Wt + (size_t)2*D_*D_, D_, D_);
    k_tr_cvt<<<trg, trb, 0, stream>>>(wo, wot, D_, D_);
    k_tr_cvt<<<trg, trb, 0, stream>>>(w1, w1t, D_, D_);
    k_tr_cvt<<<trg, trb, 0, stream>>>(w2, w2t, D_, D_);

    // 2) fused QKV projection: [16384,512] x [512,1536] -> bf16 [16384][1536]
    k_gemm<1><<<dim3(12, 128, 1), 256, 0, stream>>>(
        xb, Wt, QKV, nullptr, nullptr, BS, 3 * D_, D_, D_, D_, 3 * D_, 0, 0, 0);

    // 3) V -> V^T per batch: [2048,512](ld 1536) -> [512,2048]
    k_tr_bf16<<<dim3(16, 64, B_), trb, 0, stream>>>(
        QKV + 2 * D_, Vt, 3 * D_, S_, (long long)S_ * 3 * D_, (long long)D_ * S_);

    // 4) scores = Q K^T (raw fp32), per batch
    k_gemm<0><<<dim3(16, 16, B_), 256, 0, stream>>>(
        QKV /*Q*/, QKV + D_ /*K*/, scores, nullptr, nullptr,
        S_, S_, D_, 3 * D_, 3 * D_, S_,
        (long long)S_ * 3 * D_, (long long)S_ * 3 * D_, (long long)S_ * S_);

    // 5) softmax with 1/sqrt(DK) folded in -> bf16 attn
    k_softmax<<<BS, 256, 0, stream>>>(scores, attn, S_, 0.04419417382415922f);

    // 6) ctx = attn @ V   (B-operand = V^T)
    k_gemm<1><<<dim3(4, 16, B_), 256, 0, stream>>>(
        attn, Vt, ctx, nullptr, nullptr, S_, D_, S_, S_, S_, D_,
        (long long)S_ * S_, (long long)D_ * S_, (long long)S_ * D_);

    // 7) t1 = ctx @ wo^T + bo + x (residual fused)
    k_gemm<2><<<dim3(4, 128, 1), 256, 0, stream>>>(
        ctx, wot, t1, bo, x, BS, D_, D_, D_, D_, D_, 0, 0, 0);

    // 8) LN1 -> src1 (f32 + bf16)
    k_layernorm<<<BS / 8, 256, 0, stream>>>(t1, src1f, src1b, ln1g, ln1b);

    // 9) h = relu(src1 @ w1^T + b1) -> bf16
    k_gemm<3><<<dim3(4, 128, 1), 256, 0, stream>>>(
        src1b, w1t, h, b1, nullptr, BS, D_, D_, D_, D_, D_, 0, 0, 0);

    // 10) t2 = h @ w2^T + b2 + src1 (residual fused)
    k_gemm<2><<<dim3(4, 128, 1), 256, 0, stream>>>(
        h, w2t, t2, b2, src1f, BS, D_, D_, D_, D_, D_, 0, 0, 0);

    // 11) LN2 -> src output (fp32, straight into d_out)
    k_layernorm<<<BS / 8, 256, 0, stream>>>(t2, out_src, nullptr, ln2g, ln2b);

    // 12) mean pool + classifier -> logits
    k_pool_cls<<<B_, 512, 0, stream>>>(out_src, wc, bc, out_logits);

    (void)in_sizes; (void)n_in; (void)out_size; (void)ws_size;
}
